// OTEERE_55387898249786
// MI455X (gfx1250) — matrix operand order repaired and checked by reading, not compiled, not run on hardware
//
#include <hip/hip_runtime.h>
#include <math.h>
#include <stdint.h>

#define BSZ 16
#define SEQL 2048
#define NS 512
#define NS1 513
#define DIM 768
#define HID 384
#define G4 1536
#define OT_EPS_F 0.1f

typedef __attribute__((ext_vector_type(16))) __bf16 v16bf;
typedef __attribute__((ext_vector_type(8)))  __bf16 v8bf;
typedef __attribute__((ext_vector_type(8)))  float  v8f;

__device__ inline unsigned short f2bf(float f){
  union { float f; unsigned u; } x; x.f = f;
  unsigned r = (x.u + 0x7FFFu + ((x.u >> 16) & 1u)) >> 16;
  return (unsigned short)r;
}
__device__ inline unsigned pk2(float a, float b){
  return (unsigned)f2bf(a) | ((unsigned)f2bf(b) << 16);
}
__device__ inline v16bf mk16(v8bf lo, v8bf hi){
  v16bf r;
#pragma unroll
  for (int i=0;i<8;i++){ r[i]=lo[i]; r[i+8]=hi[i]; }
  return r;
}
__device__ inline v8f zero8(){ v8f z;
#pragma unroll
  for (int i=0;i<8;i++) z[i]=0.f;
  return z; }

// CDNA5 async global->LDS copy (ASYNCcnt-tracked); LDS dest is a per-lane byte
// offset relative to the wave's LDS base (our shared buffers sit at offset 0).
__device__ inline void async_ld32(unsigned lds_byte_off, const void* gaddr){
  asm volatile("global_load_async_to_lds_b32 %0, %1, off"
               :: "v"(lds_byte_off), "v"((unsigned long long)(uintptr_t)gaddr)
               : "memory");
}
__device__ inline void async_wait0(){
  asm volatile("s_wait_asynccnt 0" ::: "memory");
}

template<int NT>
__device__ inline float blkSum(float v, float* sh){
  int t=threadIdx.x; sh[t]=v; __syncthreads();
#pragma unroll
  for (int s=NT/2; s>0; s>>=1){ if (t<s) sh[t]+=sh[t+s]; __syncthreads(); }
  float r=sh[0]; __syncthreads(); return r;
}
template<int NT>
__device__ inline float blkMax(float v, float* sh){
  int t=threadIdx.x; sh[t]=v; __syncthreads();
#pragma unroll
  for (int s=NT/2; s>0; s>>=1){ if (t<s) sh[t]=fmaxf(sh[t],sh[t+s]); __syncthreads(); }
  float r=sh[0]; __syncthreads(); return r;
}

// ---------------- span max pool: ctx(16,2048,768) -> emb(16,512,768) ----------
__global__ __launch_bounds__(256) void k_spanmax(const float* __restrict__ ctx,
    const int* __restrict__ seg, const float* __restrict__ masks, float* __restrict__ emb){
  int n = blockIdx.x, b = blockIdx.y, t = threadIdx.x;
  float mx0=-1e30f, mx1=-1e30f, mx2=-1e30f;
  const int* sb = seg + b*SEQL;
  const float* cb = ctx + (long)b*SEQL*DIM;
  for (int s=0;s<SEQL;s++){
    if (sb[s]==n){
      const float* p = cb + (long)s*DIM;
      mx0 = fmaxf(mx0, p[t]);
      mx1 = fmaxf(mx1, p[t+256]);
      mx2 = fmaxf(mx2, p[t+512]);
    }
  }
  float m = masks[b*NS+n];
  float* e = emb + ((long)b*NS + n)*DIM;
  e[t]     = (m>0.f)? mx0 : 0.f;
  e[t+256] = (m>0.f)? mx1 : 0.f;
  e[t+512] = (m>0.f)? mx2 : 0.f;
}

// ---------------- sequence lengths from masks -------------------------------
__global__ __launch_bounds__(256) void k_lens(const float* __restrict__ masks, int* __restrict__ ls){
  __shared__ float sh[256];
  int b = blockIdx.x;
  float s = 0.f;
  for (int n=threadIdx.x; n<NS; n+=256) s += masks[b*NS+n];
  float tot = blkSum<256>(s, sh);
  if (threadIdx.x==0) ls[b] = (int)(tot + 0.5f);
}

// ---------------- reverse gather of emb -------------------------------------
__global__ __launch_bounds__(256) void k_revgather(const float* __restrict__ emb,
    const int* __restrict__ ls, float* __restrict__ embRev){
  int n = blockIdx.x, b = blockIdx.y, t = threadIdx.x;
  int lsb = ls[b];
  int rid = (n < lsb) ? (lsb-1-n) : n;
  const float* src = emb + ((long)b*NS + rid)*DIM;
  float* dst = embRev + ((long)b*NS + n)*DIM;
  dst[t] = src[t]; dst[t+256] = src[t+256]; dst[t+512] = src[t+512];
}

// ---------------- transpose+convert Whh (K x N f32) -> (N x K bf16) ---------
__global__ __launch_bounds__(256) void k_transp(const float* __restrict__ src,
    unsigned short* __restrict__ dst, int K, int N){
  long i = (long)blockIdx.x*256 + threadIdx.x;
  if (i < (long)K*N){
    long n = i / K, k = i % K;
    dst[i] = f2bf(src[k*(long)N + n]);
  }
}

// ---------------- tiled bf16 WMMA GEMM: C = A*B (+bias), f32 in/out ---------
// Requirements honored by every call site: K % 32 == 0, lda/ldb multiples of 4.
#define GMT 64
#define GNT 128
#define GKT 32
#define LSTR 40
__global__ __launch_bounds__(256) void k_gemm(
    const float* __restrict__ A, long lda, long sA,
    const float* __restrict__ B, long ldb, long sB, int transB,
    float* __restrict__ C, long ldc, long sC,
    const float* __restrict__ bias, int M, int N, int K)
{
  __shared__ __attribute__((aligned(16))) unsigned short As[GMT*LSTR];
  __shared__ __attribute__((aligned(16))) unsigned short Bs[GNT*LSTR];
  int bz = blockIdx.z;
  const float* Ab = A + (long)bz*sA;
  const float* Bb = B + (long)bz*sB;
  float* Cb = C + (long)bz*sC;
  int m0 = blockIdx.x*GMT, n0 = blockIdx.y*GNT;
  int t = threadIdx.x, wave = t>>5, lane = t&31;
  int mT = wave & 3, nG = wave >> 2;
  int lr = lane & 15, ksel = (lane>>4)*8, kb16 = (lane>>4)*16, msel = (lane>>4)<<3;
  v8f acc[4] = { zero8(), zero8(), zero8(), zero8() };
  int arow = t>>2, acg = (t&3)*8;     // A staging: 64 rows x 4 groups of 8
  int bkk = t & 31, bnb = (t>>5)*16;  // B staging (transB=0): 32 k x 8 groups of 16 n
  int bnl = t >> 1, bks = (t&1)*16;   // B staging (transB=1): 128 n x 2 groups of 16 k
  for (int k0=0; k0<K; k0+=GKT){
    // ---- stage A tile (vectorized f32->bf16, one branch per thread) ----
    {
      int gm = m0 + arow;
      unsigned r0=0,r1=0,r2=0,r3=0;
      if (gm < M){
        const float4* s0 = (const float4*)&Ab[(long)gm*lda + k0 + acg];
        float4 f0 = s0[0], f1 = s0[1];
        r0 = pk2(f0.x,f0.y); r1 = pk2(f0.z,f0.w);
        r2 = pk2(f1.x,f1.y); r3 = pk2(f1.z,f1.w);
      }
      uint4 q; q.x=r0; q.y=r1; q.z=r2; q.w=r3;
      *(uint4*)&As[arow*LSTR + acg] = q;
    }
    // ---- stage B tile transposed -> Bs[n][k] ----
    if (!transB){
      int gk = k0 + bkk;
      float vv[16];
      if (n0 + bnb + 16 <= N){
        const float4* s0 = (const float4*)&Bb[(long)gk*ldb + n0 + bnb];
        float4 f0=s0[0], f1=s0[1], f2=s0[2], f3=s0[3];
        vv[0]=f0.x; vv[1]=f0.y; vv[2]=f0.z; vv[3]=f0.w;
        vv[4]=f1.x; vv[5]=f1.y; vv[6]=f1.z; vv[7]=f1.w;
        vv[8]=f2.x; vv[9]=f2.y; vv[10]=f2.z; vv[11]=f2.w;
        vv[12]=f3.x; vv[13]=f3.y; vv[14]=f3.z; vv[15]=f3.w;
      } else {
#pragma unroll
        for (int i=0;i<16;i++){
          int gn = n0 + bnb + i;
          vv[i] = (gn < N) ? Bb[(long)gk*ldb + gn] : 0.f;
        }
      }
#pragma unroll
      for (int i=0;i<16;i++) Bs[(bnb+i)*LSTR + bkk] = f2bf(vv[i]);
    } else {
      int gn = n0 + bnl;
      unsigned r0=0,r1=0,r2=0,r3=0,r4=0,r5=0,r6=0,r7=0;
      if (gn < N){
        const float4* s0 = (const float4*)&Bb[(long)gn*ldb + k0 + bks];
        float4 f0=s0[0], f1=s0[1], f2=s0[2], f3=s0[3];
        r0=pk2(f0.x,f0.y); r1=pk2(f0.z,f0.w);
        r2=pk2(f1.x,f1.y); r3=pk2(f1.z,f1.w);
        r4=pk2(f2.x,f2.y); r5=pk2(f2.z,f2.w);
        r6=pk2(f3.x,f3.y); r7=pk2(f3.z,f3.w);
      }
      uint4 q0; q0.x=r0; q0.y=r1; q0.z=r2; q0.w=r3;
      uint4 q1; q1.x=r4; q1.y=r5; q1.z=r6; q1.w=r7;
      *(uint4*)&Bs[bnl*LSTR + bks] = q0;
      *(uint4*)&Bs[bnl*LSTR + bks + 8] = q1;
    }
    if (k0 + GKT < K)
      __builtin_prefetch(&Ab[(long)(m0+arow)*lda + k0 + GKT + acg], 0, 1);
    __syncthreads();
    v8bf alo = *(const v8bf*)&As[(mT*16+lr)*LSTR + ksel];
    v8bf ahi = *(const v8bf*)&As[(mT*16+lr)*LSTR + ksel + 16];
    v16bf af = mk16(alo, ahi);
#pragma unroll
    for (int i=0;i<4;i++){
      int ncl = nG*64 + i*16 + lr;
      v8bf blo = *(const v8bf*)&Bs[ncl*LSTR + ksel];
      v8bf bhi = *(const v8bf*)&Bs[ncl*LSTR + ksel + 16];
      v16bf bfr = mk16(blo, bhi);
      acc[i] = __builtin_amdgcn_wmma_f32_16x16x32_bf16(false, af, false, bfr,
                                                       (short)0, acc[i], false, false);
    }
    __syncthreads();
  }
#pragma unroll
  for (int i=0;i<4;i++){
    int gn = n0 + nG*64 + i*16 + lr;
    if (gn < N){
      float bv = bias ? bias[gn] : 0.f;
#pragma unroll
      for (int e=0;e<8;e++){
        int gm = m0 + mT*16 + msel + e;
        if (gm < M) Cb[(long)gm*ldc + gn] = acc[i][e] + bv;
      }
    }
  }
}

// ---------------- persistent LSTM scan (grid.x = direction) -----------------
__global__ __launch_bounds__(384) void k_lstm(
    const float* __restrict__ preG_f, const float* __restrict__ preG_b,
    const unsigned short* __restrict__ whhT_f, const unsigned short* __restrict__ whhT_b,
    const float* __restrict__ masks, float* __restrict__ outF, float* __restrict__ outB)
{
  int dir = blockIdx.x;
  const float* preG = dir ? preG_b : preG_f;
  const unsigned short* whhT = dir ? whhT_b : whhT_f;
  float* out = dir ? outB : outF;
  int ldo = dir ? HID : DIM;
  __shared__ __attribute__((aligned(16))) unsigned short hS[16*392];
  for (int i=threadIdx.x; i<16*392; i+=384) hS[i] = 0;
  __syncthreads();
  int wave = threadIdx.x >> 5, lane = threadIdx.x & 31;
  int jb = wave*32;
  int lr = lane & 15, ksel = (lane>>4)*8, kb16 = (lane>>4)*16, msel = (lane>>4)<<3;
  v8f cst[2], hst[2];
  cst[0]=zero8(); cst[1]=zero8(); hst[0]=zero8(); hst[1]=zero8();
  for (int t=0;t<NS;t++){
    v16bf afr[12];
#pragma unroll
    for (int kb=0; kb<12; kb++){
      const unsigned short* p = &hS[lr*392 + kb*32 + ksel];
      afr[kb] = mk16(*(const v8bf*)p, *(const v8bf*)(p+16));
    }
    __syncthreads();
#pragma unroll
    for (int ht=0; ht<2; ht++){
      int jbase = jb + ht*16;
      v8f ga[4];
#pragma unroll
      for (int g=0; g<4; g++){
        int ncol = g*HID + jbase + lr;
        v8f acc;
#pragma unroll
        for (int e=0;e<8;e++){
          int m = e + msel;
          acc[e] = preG[((long)m*NS + t)*G4 + ncol];
        }
#pragma unroll
        for (int kb=0; kb<12; kb++){
          const unsigned short* bp = &whhT[(long)ncol*HID + kb*32 + ksel];
          v16bf bfr = mk16(*(const v8bf*)bp, *(const v8bf*)(bp+16));
          acc = __builtin_amdgcn_wmma_f32_16x16x32_bf16(false, afr[kb], false, bfr,
                                                        (short)0, acc, false, false);
        }
        ga[g] = acc;
      }
#pragma unroll
      for (int e=0;e<8;e++){
        int m = e + msel;
        float mt = masks[m*NS + t];
        float iv = 1.f/(1.f+__expf(-ga[0][e]));
        float fv = 1.f/(1.f+__expf(-ga[1][e]));
        float gv = tanhf(ga[2][e]);
        float ov = 1.f/(1.f+__expf(-ga[3][e]));
        float cn = fv*cst[ht][e] + iv*gv;
        float hn = ov*tanhf(cn);
        float hne = (mt>0.f)? hn : hst[ht][e];
        float cne = (mt>0.f)? cn : cst[ht][e];
        hst[ht][e]=hne; cst[ht][e]=cne;
        out[((long)m*NS + t)*ldo + jbase + lr] = hne*mt;
      }
    }
    __syncthreads();
#pragma unroll
    for (int ht=0; ht<2; ht++)
#pragma unroll
      for (int e=0;e<8;e++){
        int m = e + msel;
        hS[m*392 + jb + ht*16 + lr] = f2bf(hst[ht][e]);
      }
    __syncthreads();
  }
}

// ---------------- un-reverse backward outputs into gcn_input[:, :, 384:] ----
__global__ __launch_bounds__(128) void k_unrev(const float* __restrict__ outB,
    const int* __restrict__ ls, float* __restrict__ gcnin){
  int n = blockIdx.x, b = blockIdx.y, t = threadIdx.x;
  int lsb = ls[b];
  int rid = (n < lsb) ? (lsb-1-n) : n;
  const float* src = outB + ((long)b*NS + rid)*HID;
  float* dst = gcnin + ((long)b*NS + n)*DIM + HID;
  dst[t] = src[t]; dst[t+128] = src[t+128]; dst[t+256] = src[t+256];
}

// ---------------- marginals: softmaxes -> logmu(512), lognu(513), offm ------
__global__ __launch_bounds__(512) void k_marg(const float* __restrict__ dep,
    const float* __restrict__ masks, const float* __restrict__ hd, const float* __restrict__ td,
    float* __restrict__ logmu, float* __restrict__ lognu, float* __restrict__ offm_out){
  __shared__ float sh[512];
  int b = blockIdx.x, n = threadIdx.x;
  float dv = dep[b*NS+n], mv = masks[b*NS+n];
  float nm = -fminf(hd[b*NS+n], td[b*NS+n]);
  float om = (1.f - dv)*mv;
  float ons  = (dv==0.f)? -1e4f : nm*dv;
  float offs = (om==0.f)? -1e4f : nm*om;
  float mx = blkMax<512>(ons, sh);
  float ex = __expf(ons - mx);
  float se = blkSum<512>(ex, sh);
  float onmarg = ex / se;
  mx = blkMax<512>(offs, sh);
  ex = __expf(offs - mx);
  se = blkSum<512>(ex, sh);
  float offmarg = ex / se;
  offm_out[b*NS+n] = om;
  logmu[b*NS+n] = logf(offmarg + 1e-8f);
  lognu[b*NS1 + 1 + n] = logf(0.5f*onmarg + 1e-8f);
  if (n==0) lognu[b*NS1] = logf(0.5f + 1e-8f);
}

// ---------------- off_dp / on_dps[1:] ---------------------------------------
__global__ __launch_bounds__(256) void k_builddp(const float* __restrict__ gin,
    const float* __restrict__ dep, const float* __restrict__ offm,
    float* __restrict__ offdp, float* __restrict__ ondps){
  int n = blockIdx.x, b = blockIdx.y, t = threadIdx.x;
  float om = offm[b*NS+n], dv = dep[b*NS+n];
#pragma unroll
  for (int i=0;i<3;i++){
    int d = t + i*256;
    long idx = ((long)b*NS + n)*DIM + d;
    float g = gin[idx];
    offdp[idx] = g*om;
    ondps[((long)b*NS1 + 1 + n)*DIM + d] = g*dv;
  }
}
__global__ __launch_bounds__(256) void k_nullp(const float* __restrict__ offdp,
    float* __restrict__ ondps){
  int b = blockIdx.x, t = threadIdx.x;
#pragma unroll
  for (int i=0;i<3;i++){
    int d = t + i*256;
    float s = 0.f;
    for (int n=0;n<NS;n++) s += offdp[((long)b*NS + n)*DIM + d];
    ondps[(long)b*NS1*DIM + d] = s / (float)NS;
  }
}

// ---------------- row squared norms -----------------------------------------
__global__ __launch_bounds__(256) void k_rownorm(const float* __restrict__ X,
    float* __restrict__ out, int cols){
  __shared__ float sh[256];
  long r = blockIdx.x;
  float a = 0.f;
  for (int d=threadIdx.x; d<cols; d+=256){ float v = X[r*cols + d]; a += v*v; }
  float s = blkSum<256>(a, sh);
  if (threadIdx.x==0) out[r] = s;
}

// ---------------- C = xx + yy - 2*dot (in place) ----------------------------
__global__ __launch_bounds__(256) void k_fixC(float* __restrict__ C,
    const float* __restrict__ xx, const float* __restrict__ yy){
  int n = blockIdx.x, b = blockIdx.y;
  float xv = xx[b*NS+n];
  for (int m=threadIdx.x; m<NS1; m+=256){
    long idx = ((long)b*NS + n)*NS1 + m;
    C[idx] = xv + yy[b*NS1+m] - 2.f*C[idx];
  }
}

__global__ __launch_bounds__(256) void k_zero(float* __restrict__ p, long n){
  long i = (long)blockIdx.x*256 + threadIdx.x;
  if (i < n) p[i] = 0.f;
}

// ---------------- Sinkhorn updates (async global->LDS staging of C) --------
__global__ __launch_bounds__(256) void k_sinku(const float* __restrict__ C,
    const float* __restrict__ v, const float* __restrict__ logmu, float* __restrict__ u){
  __shared__ __attribute__((aligned(16))) float smem[520 + 256];
  float* row = smem;            // C row cache (LDS offset 0)
  float* red = smem + 520;      // reduction scratch
  int n = blockIdx.x, b = blockIdx.y;
  const float* Cr = C + ((long)b*NS + n)*NS1;
  const float* vb = v + b*NS1;
  int t = threadIdx.x;
  async_ld32((unsigned)(t*4),        Cr + t);
  async_ld32((unsigned)((t+256)*4),  Cr + t + 256);
  if (t==0) async_ld32((unsigned)(512*4), Cr + 512);
  async_wait0();
  __syncthreads();
  float ub = u[b*NS+n];
  float mx = -3e38f;
  for (int m=t; m<NS1; m+=256) mx = fmaxf(mx, vb[m] - row[m]);
  float MX = blkMax<256>(mx, red);
  float s = 0.f;
  for (int m=t; m<NS1; m+=256) s += __expf((vb[m]-row[m]-MX)*(1.f/OT_EPS_F));
  float S = blkSum<256>(s, red);
  if (t==0){
    float lse = (MX + ub)*(1.f/OT_EPS_F) + logf(S);
    u[b*NS+n] = ub + OT_EPS_F*(logmu[b*NS+n] - lse);
  }
}
__global__ __launch_bounds__(256) void k_sinkv(const float* __restrict__ C,
    const float* __restrict__ u, const float* __restrict__ lognu, float* __restrict__ v){
  __shared__ __attribute__((aligned(16))) float smem[512 + 256];
  float* col = smem;            // C column cache (LDS offset 0)
  float* red = smem + 512;
  int m = blockIdx.x, b = blockIdx.y;
  const float* ub = u + b*NS;
  int t = threadIdx.x;
  async_ld32((unsigned)(t*4),       &C[((long)b*NS + t)*NS1 + m]);
  async_ld32((unsigned)((t+256)*4), &C[((long)b*NS + t + 256)*NS1 + m]);
  async_wait0();
  __syncthreads();
  float vb = v[b*NS1+m];
  float mx = -3e38f;
  for (int n=t; n<NS; n+=256) mx = fmaxf(mx, ub[n] - col[n]);
  float MX = blkMax<256>(mx, red);
  float s = 0.f;
  for (int n=t; n<NS; n+=256) s += __expf((ub[n] - col[n] - MX)*(1.f/OT_EPS_F));
  float S = blkSum<256>(s, red);
  if (t==0){
    float lse = (MX + vb)*(1.f/OT_EPS_F) + logf(S);
    v[b*NS1+m] = vb + OT_EPS_F*(lognu[b*NS1+m] - lse);
  }
}

// ---------------- pi, pimax --------------------------------------------------
__global__ __launch_bounds__(256) void k_pi(const float* __restrict__ C,
    const float* __restrict__ u, const float* __restrict__ v,
    float* __restrict__ pi, float* __restrict__ pimax){
  __shared__ float sh[256];
  int n = blockIdx.x, b = blockIdx.y;
  float ub = u[b*NS+n];
  float mx = -3e38f;
  for (int m=threadIdx.x; m<NS1; m+=256){
    long idx = ((long)b*NS + n)*NS1 + m;
    float p = __expf((-C[idx] + ub + v[b*NS1+m])*(1.f/OT_EPS_F));
    pi[idx] = p;
    mx = fmaxf(mx, p);
  }
  float M = blkMax<256>(mx, sh);
  if (threadIdx.x==0) pimax[b*NS+n] = M;
}

// ---------------- pruned adjacency ------------------------------------------
__global__ __launch_bounds__(512) void k_pruned(const float* __restrict__ pi,
    const float* __restrict__ pimax, const float* __restrict__ dep,
    const float* __restrict__ offm, const float* __restrict__ adjs, float* __restrict__ pruned){
  int n = blockIdx.x, b = blockIdx.y, j = threadIdx.x;
  float o1 = (pi[((long)b*NS + n)*NS1 + j + 1] >= pimax[b*NS+n]) ? dep[b*NS+j]*offm[b*NS+n] : 0.f;
  float o2 = (pi[((long)b*NS + j)*NS1 + n + 1] >= pimax[b*NS+j]) ? dep[b*NS+n]*offm[b*NS+j] : 0.f;
  float da = adjs[((long)b*NS + n)*NS + j]*dep[b*NS+n]*dep[b*NS+j];
  pruned[((long)b*NS + n)*NS + j] = (da + o1 + o2 > 0.f) ? 1.f : 0.f;
}

// ---------------- GCN helpers ------------------------------------------------
__global__ __launch_bounds__(256) void k_denom(const float* __restrict__ adj, float* __restrict__ denom){
  __shared__ float sh[256];
  int n = blockIdx.x, b = blockIdx.y;
  float s = 0.f;
  for (int j=threadIdx.x; j<NS; j+=256) s += adj[((long)b*NS + n)*NS + j];
  float S = blkSum<256>(s, sh);
  if (threadIdx.x==0) denom[b*NS+n] = S + 1.f;
}
__global__ __launch_bounds__(256) void k_addv(const float* __restrict__ a,
    const float* __restrict__ b, float* __restrict__ c, long n){
  long i = (long)blockIdx.x*256 + threadIdx.x;
  if (i < n) c[i] = a[i] + b[i];
}
__global__ __launch_bounds__(256) void k_postgcn(const float* __restrict__ G,
    const float* __restrict__ denom, float* __restrict__ h){
  int n = blockIdx.x, b = blockIdx.y, t = threadIdx.x;
  float dn = denom[b*NS+n];
#pragma unroll
  for (int i=0;i<3;i++){
    long idx = ((long)b*NS + n)*DIM + t + i*256;
    float v = G[idx]/dn;
    h[idx] = v > 0.f ? v : 0.f;
  }
}

// ---------------- pooling (heads|tails|doc -> pres), full_doc ---------------
__global__ __launch_bounds__(256) void k_pool(const float* __restrict__ gout,
    const float* __restrict__ gin, const int* __restrict__ spans, float* __restrict__ pres){
  int b = blockIdx.x, t = threadIdx.x;
  int s0 = spans[b*4+0], e0 = spans[b*4+1], s1 = spans[b*4+2], e1 = spans[b*4+3];
#pragma unroll
  for (int i=0;i<3;i++){
    int d = t + i*256;
    float hm=-1e9f, tm=-1e9f, dm=-3.0e38f;
    for (int n=0;n<NS;n++){
      long idx = ((long)b*NS + n)*DIM + d;
      float r = gout[idx] + gin[idx];
      if (n>=s0 && n<=e0) hm = fmaxf(hm, r);
      if (n>=s1 && n<=e1) tm = fmaxf(tm, r);
      dm = fmaxf(dm, r);
    }
    pres[b*2304 + d] = hm;
    pres[b*2304 + 768 + d] = tm;
    pres[b*2304 + 1536 + d] = dm;
  }
}
__global__ __launch_bounds__(256) void k_fulldoc(const float* __restrict__ gout,
    const float* __restrict__ gin, float* __restrict__ fulldoc){
  int b = blockIdx.x, t = threadIdx.x;
#pragma unroll
  for (int i=0;i<3;i++){
    int d = t + i*256;
    float dm = -3.0e38f;
    for (int n=0;n<NS;n++){
      long idx = ((long)b*NS + n)*DIM + d;
      dm = fmaxf(dm, gout[idx] + gin[idx]);
    }
    fulldoc[b*DIM + d] = dm;
  }
}

__global__ __launch_bounds__(256) void k_relu(float* __restrict__ p, long n){
  long i = (long)blockIdx.x*256 + threadIdx.x;
  if (i < n && p[i] < 0.f) p[i] = 0.f;
}

// ---------------- final loss + output copy ----------------------------------
__global__ __launch_bounds__(256) void k_loss(const float* __restrict__ logits,
    const int* __restrict__ labels, const float* __restrict__ lossw,
    const float* __restrict__ pres, const float* __restrict__ fulldoc, float* __restrict__ dout){
  __shared__ float sh[256];
  int t = threadIdx.x;
  if (t < 64) dout[t] = logits[t];
  float ce_num = 0.f, wsum = 0.f;
  if (t < BSZ){
    const float* l = logits + t*4;
    float m = fmaxf(fmaxf(l[0],l[1]), fmaxf(l[2],l[3]));
    float s = __expf(l[0]-m)+__expf(l[1]-m)+__expf(l[2]-m)+__expf(l[3]-m);
    float lse = m + logf(s);
    int lab = labels[t];
    float w = lossw[lab];
    ce_num = w * (l[lab] - lse);
    wsum = w;
  }
  float totCe = blkSum<256>(ce_num, sh);
  float totW  = blkSum<256>(wsum, sh);
  float acc = 0.f;
  for (long i=t; i<(long)BSZ*DIM; i+=256){
    long b = i / DIM, d = i % DIM;
    float diff = pres[b*2304 + 1536 + d] - fulldoc[i];
    acc += diff*diff;
  }
  float totM = blkSum<256>(acc, sh);
  if (t==0) dout[64] = (-totCe/totW) + 0.1f*(totM/(float)((long)BSZ*DIM));
}

// ============================================================================
extern "C" void kernel_launch(void* const* d_in, const int* in_sizes, int n_in,
                              void* d_out, int out_size, void* d_ws, size_t ws_size,
                              hipStream_t stream)
{
  (void)in_sizes; (void)n_in; (void)out_size; (void)ws_size;
  const float* ctx   = (const float*)d_in[0];
  const float* masks = (const float*)d_in[1];
  const float* dep   = (const float*)d_in[2];
  const float* hd    = (const float*)d_in[3];
  const float* td    = (const float*)d_in[4];
  const float* adjs  = (const float*)d_in[5];
  const float* Wih_f = (const float*)d_in[6];
  const float* Whh_f = (const float*)d_in[7];
  const float* b_f   = (const float*)d_in[8];
  const float* Wih_b = (const float*)d_in[9];
  const float* Whh_b = (const float*)d_in[10];
  const float* b_b   = (const float*)d_in[11];
  const float* gcnW  = (const float*)d_in[12];
  const float* gcnb  = (const float*)d_in[13];
  const float* fc1W  = (const float*)d_in[14];
  const float* fc1b  = (const float*)d_in[15];
  const float* fc2W  = (const float*)d_in[16];
  const float* fc2b  = (const float*)d_in[17];
  const float* lossw = (const float*)d_in[18];
  const int* seg     = (const int*)d_in[19];
  const int* labels  = (const int*)d_in[20];
  const int* spans   = (const int*)d_in[21];

  const long N_emb   = (long)BSZ*NS*DIM;
  const long N_preG  = (long)BSZ*NS*G4;
  const long N_C     = (long)BSZ*NS*NS1;
  const long N_adjE  = (long)BSZ*NS*NS;
  const long N_outB  = (long)BSZ*NS*HID;
  const long N_ondps = (long)BSZ*NS1*DIM;

  char* W = (char*)d_ws;
  size_t o = 0;
  auto alloc = [&](size_t bytes){ size_t r = o; o = (o + bytes + 255) & ~(size_t)255; return r; };
  int*   ls      = (int*)  (W + alloc(16*sizeof(int)));
  float* logmu   = (float*)(W + alloc((size_t)BSZ*NS*4));
  float* lognu   = (float*)(W + alloc((size_t)BSZ*NS1*4));
  float* offm    = (float*)(W + alloc((size_t)BSZ*NS*4));
  float* u       = (float*)(W + alloc((size_t)BSZ*NS*4));
  float* v       = (float*)(W + alloc((size_t)BSZ*NS1*4));
  float* pimax   = (float*)(W + alloc((size_t)BSZ*NS*4));
  float* xx      = (float*)(W + alloc((size_t)BSZ*NS*4));
  float* yy      = (float*)(W + alloc((size_t)BSZ*NS1*4));
  float* denom   = (float*)(W + alloc((size_t)BSZ*NS*4));
  float* pres    = (float*)(W + alloc((size_t)BSZ*2304*4));
  float* fc1out  = (float*)(W + alloc((size_t)BSZ*1152*4));
  float* logits  = (float*)(W + alloc((size_t)BSZ*4*4));
  float* fulldoc = (float*)(W + alloc((size_t)BSZ*DIM*4));
  unsigned short* whhTf = (unsigned short*)(W + alloc((size_t)G4*HID*2));
  unsigned short* whhTb = (unsigned short*)(W + alloc((size_t)G4*HID*2));
  float* gcnin   = (float*)(W + alloc((size_t)N_emb*4));
  float* outB    = (float*)(W + alloc((size_t)N_outB*4));
  size_t big     = alloc(0);
  // phase1 overlay
  float* emb    = (float*)(W + big);
  float* embRev = emb + N_emb;
  float* preGf  = embRev + N_emb;
  float* preGb  = preGf + N_preG;
  // phase2 overlay (live only after the LSTM completes)
  float* offdp  = (float*)(W + big);
  float* ondps  = offdp + N_emb;
  float* Cmat   = ondps + N_ondps;
  float* pim    = Cmat + N_C;        // pi
  float* pruned = pim + N_C;
  // phase3 overlay (live only after pruned is built / consumed)
  float* Ax     = (float*)(W + big);
  float* Sb     = Ax + N_emb;
  float* h1     = Sb + N_emb;
  float* fullout= h1 + N_emb;        // used after GCN#1 (pi/pruned dead)
  float* gcnout = pruned + N_adjE;   // beyond pruned, survives pooling

  auto gemm = [&](const float* A, long lda, long sA, const float* B, long ldb, long sB,
                  int tB, float* C, long ldc, long sC, const float* bias,
                  int M, int N, int K, int batch){
    dim3 g((M+GMT-1)/GMT, (N+GNT-1)/GNT, batch);
    k_gemm<<<g, dim3(256), 0, stream>>>(A, lda, sA, B, ldb, sB, tB, C, ldc, sC, bias, M, N, K);
  };

  // ---- encoder ----
  k_spanmax<<<dim3(NS,BSZ), 256, 0, stream>>>(ctx, seg, masks, emb);
  k_lens<<<BSZ, 256, 0, stream>>>(masks, ls);
  k_revgather<<<dim3(NS,BSZ), 256, 0, stream>>>(emb, ls, embRev);
  {
    long nel = (long)G4*HID;
    k_transp<<<(unsigned)((nel+255)/256), 256, 0, stream>>>(Whh_f, whhTf, HID, G4);
    k_transp<<<(unsigned)((nel+255)/256), 256, 0, stream>>>(Whh_b, whhTb, HID, G4);
  }
  gemm(emb,    DIM, 0, Wih_f, G4, 0, 0, preGf, G4, 0, b_f, BSZ*NS, G4, DIM, 1);
  gemm(embRev, DIM, 0, Wih_b, G4, 0, 0, preGb, G4, 0, b_b, BSZ*NS, G4, DIM, 1);
  k_lstm<<<2, 384, 0, stream>>>(preGf, preGb, whhTf, whhTb, masks, gcnin, outB);
  k_unrev<<<dim3(NS,BSZ), 128, 0, stream>>>(outB, ls, gcnin);

  // ---- OT inputs ----
  k_marg<<<BSZ, 512, 0, stream>>>(dep, masks, hd, td, logmu, lognu, offm);
  k_builddp<<<dim3(NS,BSZ), 256, 0, stream>>>(gcnin, dep, offm, offdp, ondps);
  k_nullp<<<BSZ, 256, 0, stream>>>(offdp, ondps);
  k_rownorm<<<BSZ*NS,  256, 0, stream>>>(offdp, xx, DIM);
  k_rownorm<<<BSZ*NS1, 256, 0, stream>>>(ondps, yy, DIM);
  gemm(offdp, DIM, (long)NS*DIM, ondps, DIM, (long)NS1*DIM, 1,
       Cmat, NS1, (long)NS*NS1, nullptr, NS, NS1, DIM, BSZ);
  k_fixC<<<dim3(NS,BSZ), 256, 0, stream>>>(Cmat, xx, yy);

  // ---- Sinkhorn ----
  k_zero<<<(BSZ*NS+255)/256, 256, 0, stream>>>(u, BSZ*NS);
  k_zero<<<(BSZ*NS1+255)/256, 256, 0, stream>>>(v, BSZ*NS1);
  for (int it=0; it<100; ++it){
    k_sinku<<<dim3(NS,BSZ),  256, 0, stream>>>(Cmat, v, logmu, u);
    k_sinkv<<<dim3(NS1,BSZ), 256, 0, stream>>>(Cmat, u, lognu, v);
  }
  k_pi<<<dim3(NS,BSZ), 256, 0, stream>>>(Cmat, u, v, pim, pimax);
  k_pruned<<<dim3(NS,BSZ), 512, 0, stream>>>(pim, pimax, dep, offm, adjs, pruned);

  // ---- GCN (shared driver) ----
  auto gcnrun = [&](const float* adj, float* outPtr){
    k_denom<<<dim3(NS,BSZ), 256, 0, stream>>>(adj, denom);
    const float* cur = gcnin;
    float* outs[2] = { h1, outPtr };
    for (int l=0; l<2; l++){
      gemm(adj, NS, (long)NS*NS, cur, DIM, (long)NS*DIM, 0,
           Ax, DIM, (long)NS*DIM, nullptr, NS, DIM, NS, BSZ);
      k_addv<<<(unsigned)((N_emb+255)/256), 256, 0, stream>>>(Ax, cur, Sb, N_emb);
      gemm(Sb, DIM, (long)NS*DIM, gcnW + (long)l*DIM*DIM, DIM, 0, 0,
           Ax, DIM, (long)NS*DIM, gcnb + l*DIM, NS, DIM, DIM, BSZ);
      k_postgcn<<<dim3(NS,BSZ), 256, 0, stream>>>(Ax, denom, outs[l]);
      cur = outs[l];
    }
  };
  gcnrun(pruned, gcnout);
  k_pool<<<BSZ, 256, 0, stream>>>(gcnout, gcnin, spans, pres);
  gcnrun(adjs, fullout);
  k_fulldoc<<<BSZ, 256, 0, stream>>>(fullout, gcnin, fulldoc);

  // ---- classifier head + loss ----
  gemm(pres,   2304, 0, fc1W, 1152, 0, 0, fc1out, 1152, 0, fc1b, BSZ, 1152, 2304, 1);
  k_relu<<<(BSZ*1152+255)/256, 256, 0, stream>>>(fc1out, BSZ*1152);
  gemm(fc1out, 1152, 0, fc2W, 4, 0, 0, logits, 4, 0, fc2b, BSZ, 4, 1152, 1);
  k_loss<<<1, 256, 0, stream>>>(logits, labels, lossw, pres, fulldoc, (float*)d_out);
}
